// MultiModalModel_54065048322130
// MI455X (gfx1250) — matrix operand order repaired and checked by reading, not compiled, not run on hardware
//
#include <hip/hip_runtime.h>
#include <math.h>

// ---------------- problem constants ----------------
constexpr int NB   = 4;      // batch
constexpr int DM   = 64;     // d_model
constexpr int HH   = 64;
constexpr int WW   = 64;
constexpr int LSEQ = HH*WW;  // 4096
constexpr int DI   = 128;    // d_inner
constexpr int DST  = 16;     // d_state
constexpr int KD   = 4;      // directions

typedef __attribute__((ext_vector_type(16))) _Float16 v16h;
typedef __attribute__((ext_vector_type(8)))  float    v8f;

// ---------------- WMMA helpers (CDNA5 16x16x32 f16 -> f32) ----------------
__device__ __forceinline__ v8f wmma16(v16h a, v16h b, v8f c) {
  // D = A(16x32 f16) x B(32x16 f16) + C(16x16 f32)
  return __builtin_amdgcn_wmma_f32_16x16x32_f16(false, a, false, b, (short)0, c,
                                                false, false);
}

// A fragment: 16x32 f16: lane m=lane&15; K = i + 16*(lane>=16) (matches B).
// element i: K = i + 16*(lane>=16)
__device__ __forceinline__ v16h load_a_frag(const _Float16* a, int ldk, int kbase,
                                            int lane) {
  int m  = lane & 15;
  int kg = (lane >> 4) << 4;
  const _Float16* p = a + m * ldk + kbase;
  v16h r;
#pragma unroll
  for (int i = 0; i < 16; ++i) {
    int j  = i >> 1;
    int kk = i + kg + (j & 0);
    r[i] = p[kk];
  }
  return r;
}

// B fragment: 32x16 f16 from row-major f32 weights W[n][k] (row stride ldk).
// lane n=lane&15 holds column n, K = kbase + 16*(lane>=16) + i  (contiguous run
// of 16 -> vectorizes to b128 global loads). Rows >= nmax are zero-padded.
__device__ __forceinline__ v16h load_b_frag(const float* __restrict__ W, int ldk,
                                            int n0, int kbase, int lane, int nmax) {
  int n = n0 + (lane & 15);
  v16h r;
  if (n < nmax) {
    const float* p = W + (size_t)n * ldk + kbase + ((lane >> 4) << 4);
#pragma unroll
    for (int i = 0; i < 16; ++i) r[i] = (_Float16)p[i];
  } else {
#pragma unroll
    for (int i = 0; i < 16; ++i) r[i] = (_Float16)0.f;
  }
  return r;
}

__device__ __forceinline__ float silu_f(float x) { return x / (1.f + __expf(-x)); }

// cross-scan source spatial index for direction k at sequence step t
__device__ __forceinline__ int scan_src(int k, int t) {
  int u = (k & 2) ? (LSEQ - 1 - t) : t;
  if (k & 1) u = ((u & 63) << 6) | (u >> 6);  // H<->W transpose (64x64)
  return u;
}

// =====================================================================
// K1: LN(x0), LN(x1) -> input projections (two GEMMs, N=256,K=64), split
//     into proj-sum (d<128, linear) and z-sum (silu each, then add).
// block = 512 (16 waves), one wave per 16-col tile; 16 seq positions/block
// =====================================================================
__global__ __launch_bounds__(512) void k_inproj(
    const float* __restrict__ x0, const float* __restrict__ x1,
    const float* __restrict__ g, const float* __restrict__ be,
    const float* __restrict__ W0, const float* __restrict__ W1,
    float* __restrict__ xbuf /*(B,128,L)*/, float* __restrict__ zbuf /*(B,L,128)*/) {
  __shared__ _Float16 a0[16][64];
  __shared__ _Float16 a1[16][64];
  int blk = blockIdx.x;               // NB * (LSEQ/16)
  int b   = blk >> 8;
  int l0  = (blk & 255) << 4;
  int tid = threadIdx.x;
  int pos = tid >> 5, lane = tid & 31;
  int l = l0 + pos;
  { // layernorm over 64 channels, 32 lanes x 2 channels each, both inputs
    int c0 = lane << 1;
    size_t base = ((size_t)b * DM + c0) * LSEQ + l;
    float v0 = x0[base], v1 = x0[base + LSEQ];
    float u0 = x1[base], u1 = x1[base + LSEQ];
    float s = v0 + v1, sq = v0 * v0 + v1 * v1;
    float s2 = u0 + u1, sq2 = u0 * u0 + u1 * u1;
#pragma unroll
    for (int o = 16; o > 0; o >>= 1) {
      s += __shfl_xor(s, o, 32);   sq += __shfl_xor(sq, o, 32);
      s2 += __shfl_xor(s2, o, 32); sq2 += __shfl_xor(sq2, o, 32);
    }
    float m0 = s / 64.f,  r0 = rsqrtf(sq / 64.f - m0 * m0 + 1e-5f);
    float m1 = s2 / 64.f, r1 = rsqrtf(sq2 / 64.f - m1 * m1 + 1e-5f);
    a0[pos][c0]     = (_Float16)((v0 - m0) * r0 * g[c0] + be[c0]);
    a0[pos][c0 + 1] = (_Float16)((v1 - m0) * r0 * g[c0 + 1] + be[c0 + 1]);
    a1[pos][c0]     = (_Float16)((u0 - m1) * r1 * g[c0] + be[c0]);
    a1[pos][c0 + 1] = (_Float16)((u1 - m1) * r1 * g[c0 + 1] + be[c0 + 1]);
  }
  __syncthreads();
  int n0 = (tid >> 5) << 4;  // wave -> 16-col tile of the 256 outputs
  v8f acc0 = {}, acc1 = {};
#pragma unroll
  for (int ks = 0; ks < 64; ks += 32) {
    v16h Af0 = load_a_frag(&a0[0][0], 64, ks, lane);
    v16h Bf0 = load_b_frag(W0, 64, n0, ks, lane, 256);
    acc0 = wmma16(Af0, Bf0, acc0);
    v16h Af1 = load_a_frag(&a1[0][0], 64, ks, lane);
    v16h Bf1 = load_b_frag(W1, 64, n0, ks, lane, 256);
    acc1 = wmma16(Af1, Bf1, acc1);
  }
  int n = n0 + (lane & 15);
  int mb = (lane >> 4) << 3;
  if (n < DI) {  // proj sum, write (B,128,L)
    float* dst = xbuf + ((size_t)(b * DI + n)) * LSEQ + l0 + mb;
#pragma unroll
    for (int v = 0; v < 8; ++v) dst[v] = acc0[v] + acc1[v];
  } else {       // z = silu(p0)+silu(p1), write (B,L,128)
    int d = n - DI;
    float* dst = zbuf + ((size_t)b * LSEQ + l0 + mb) * DI + d;
#pragma unroll
    for (int v = 0; v < 8; ++v)
      dst[(size_t)v * DI] = silu_f(acc0[v]) + silu_f(acc1[v]);
  }
}

// =====================================================================
// K2: per-direction x_dbl = xs @ x_proj_w^T (N=36,K=128) + dt projection
//     (+dt_bias, softplus). xs gathered on the fly (no materialization).
// block = 128 (4 waves): waves 0..2 cover c-tiles 0..47 (rows>=36 zero)
// =====================================================================
__global__ __launch_bounds__(128) void k_xdbl(
    const float* __restrict__ xbuf, const float* __restrict__ xpw,
    const float* __restrict__ dtw, const float* __restrict__ dtb,
    float* __restrict__ BCbuf /*(B,K,L,32)*/, float* __restrict__ dtbuf /*(B,K,L,128)*/) {
  __shared__ _Float16 axs[16][128];
  __shared__ float dt4[16][4];
  int blk = blockIdx.x;                 // NB*KD*(LSEQ/16)
  int t0 = (blk & 255) << 4;
  int k  = (blk >> 8) & 3;
  int b  = blk >> 10;
  int tid = threadIdx.x;
  { // stage A: 16 steps x 128 channels, gathered per direction
    const float* src = xbuf + ((size_t)(b * DI + tid)) * LSEQ;
#pragma unroll
    for (int m = 0; m < 16; ++m) axs[m][tid] = (_Float16)src[scan_src(k, t0 + m)];
  }
  __syncthreads();
  int w = tid >> 5, lane = tid & 31;
  int c0 = w << 4;
  if (c0 < 48) {
    const float* Wk = xpw + (size_t)k * 36 * DI;
    v8f acc = {};
#pragma unroll
    for (int ks = 0; ks < DI; ks += 32) {
      v16h Af = load_a_frag(&axs[0][0], DI, ks, lane);
      v16h Bf = load_b_frag(Wk, DI, c0, ks, lane, 36);
      acc = wmma16(Af, Bf, acc);
    }
    int c = c0 + (lane & 15);
    int mb = (lane >> 4) << 3;
    if (c < 4) {
#pragma unroll
      for (int v = 0; v < 8; ++v) dt4[mb + v][c] = acc[v];
    } else if (c < 36) {  // B rows at 0..15, C rows at 16..31 (= c-4)
      float* dst = BCbuf + ((size_t)(b * KD + k) * LSEQ + t0 + mb) * 32 + (c - 4);
#pragma unroll
      for (int v = 0; v < 8; ++v) dst[(size_t)v * 32] = acc[v];
    }
  }
  __syncthreads();
  { // dt projection: d = tid, K=4; + bias; softplus
    float wr[4];
#pragma unroll
    for (int r = 0; r < 4; ++r) wr[r] = dtw[((size_t)k * DI + tid) * 4 + r];
    float bias = dtb[k * DI + tid];
    float* ddst = dtbuf + ((size_t)(b * KD + k) * LSEQ + t0) * DI + tid;
#pragma unroll
    for (int m = 0; m < 16; ++m) {
      float v = bias;
#pragma unroll
      for (int r = 0; r < 4; ++r) v += dt4[m][r] * wr[r];
      ddst[(size_t)m * DI] = (v > 20.f) ? v : log1pf(__expf(v));
    }
  }
}

// =====================================================================
// K3: selective scan. lane = (d%2 across half-waves) x (n=lane&15).
//     h kept in a register; y = shfl-reduction over the 16 state lanes.
// block = 512 (16 waves = 32 channels); grid = B*K*4
// =====================================================================
__global__ __launch_bounds__(512) void k_scan(
    const float* __restrict__ xbuf, const float* __restrict__ dtbuf,
    const float* __restrict__ BCbuf, const float* __restrict__ Alogs,
    const float* __restrict__ Dsv, float* __restrict__ ybuf /*(B,K,128,L)*/) {
  int blk = blockIdx.x;
  int dg = blk & 3;
  int k  = (blk >> 2) & 3;
  int b  = blk >> 4;
  int tid = threadIdx.x;
  int w = tid >> 5, lane = tid & 31;
  int n = lane & 15;
  int d = dg * 32 + w * 2 + (lane >> 4);
  float Aval = -__expf(Alogs[((size_t)(k * DI + d)) * DST + n]);
  float Dval = Dsv[k * DI + d];
  size_t bk = (size_t)(b * KD + k);
  const float* xsrc  = xbuf + ((size_t)(b * DI + d)) * LSEQ;
  const float* dsrc  = dtbuf + bk * LSEQ * DI + d;
  const float* bcsrc = BCbuf + bk * LSEQ * 32;
  float* ydst = ybuf + (bk * DI + d) * LSEQ;
  float h = 0.f;
  for (int t = 0; t < LSEQ; ++t) {
    float dtv = dsrc[(size_t)t * DI];
    float xv  = xsrc[scan_src(k, t)];
    float Bn  = bcsrc[(size_t)t * 32 + n];
    float Cn  = bcsrc[(size_t)t * 32 + 16 + n];
    h = __expf(dtv * Aval) * h + (dtv * xv) * Bn;
    float p = h * Cn;
#pragma unroll
    for (int o = 8; o > 0; o >>= 1) p += __shfl_xor(p, o, 32);
    if (n == 0) ydst[t] = p + Dval * xv;
    if (t + 1 < LSEQ) {  // hide next-step latency (global_prefetch_b8)
      __builtin_prefetch(&bcsrc[(size_t)(t + 1) * 32], 0, 1);
      __builtin_prefetch(&dsrc[(size_t)(t + 1) * DI], 0, 1);
    }
  }
}

// =====================================================================
// K4: merge 4 directions (gather with flips/transposes), LN over 128,
//     multiply by z gate, emit f16 activations. block=128, one (b,l)/block
// =====================================================================
__global__ __launch_bounds__(128) void k_combine(
    const float* __restrict__ ybuf, const float* __restrict__ zbuf,
    const float* __restrict__ g, const float* __restrict__ be,
    _Float16* __restrict__ yz /*(B*L,128)*/) {
  int blk = blockIdx.x;       // NB*LSEQ
  int l = blk & (LSEQ - 1);
  int b = blk / LSEQ;
  int d = threadIdx.x;
  int t1 = ((l & 63) << 6) | (l >> 6);
  size_t base = (size_t)b * KD * DI * LSEQ;
  float v = ybuf[base + ((size_t)(0 * DI + d)) * LSEQ + l]
          + ybuf[base + ((size_t)(2 * DI + d)) * LSEQ + (LSEQ - 1 - l)]
          + ybuf[base + ((size_t)(1 * DI + d)) * LSEQ + t1]
          + ybuf[base + ((size_t)(3 * DI + d)) * LSEQ + (LSEQ - 1 - t1)];
  __shared__ float red[8];
  float s = v, sq = v * v;
#pragma unroll
  for (int o = 16; o > 0; o >>= 1) { s += __shfl_xor(s, o, 32); sq += __shfl_xor(sq, o, 32); }
  int w = threadIdx.x >> 5, lane = threadIdx.x & 31;
  if (lane == 0) { red[w] = s; red[4 + w] = sq; }
  __syncthreads();
  s  = red[0] + red[1] + red[2] + red[3];
  sq = red[4] + red[5] + red[6] + red[7];
  float mean = s / 128.f;
  float rs = rsqrtf(sq / 128.f - mean * mean + 1e-5f);
  float y = (v - mean) * rs * g[d] + be[d];
  size_t idx = ((size_t)b * LSEQ + l) * DI + d;
  yz[idx] = (_Float16)(y * zbuf[idx]);
}

// =====================================================================
// K5: out-projection (N=64,K=128) + residual (x0+x1 gathered transposed)
// block = 128 (4 waves x 16-col tiles), 16 rows/block
// =====================================================================
__global__ __launch_bounds__(128) void k_outproj(
    const _Float16* __restrict__ yz, const float* __restrict__ Wo,
    const float* __restrict__ x0, const float* __restrict__ x1,
    _Float16* __restrict__ fus /*(B*L,64)*/) {
  __shared__ _Float16 asb[16][128];
  int r0 = blockIdx.x << 4;
  int tid = threadIdx.x;
  const _Float16* src = yz + (size_t)r0 * DI;
#pragma unroll
  for (int i = 0; i < 16; ++i) ((_Float16*)asb)[tid + i * 128] = src[tid + i * 128];
  __syncthreads();
  int w = tid >> 5, lane = tid & 31;
  int n0 = w << 4;
  v8f acc = {};
#pragma unroll
  for (int ks = 0; ks < DI; ks += 32) {
    v16h Af = load_a_frag(&asb[0][0], DI, ks, lane);
    v16h Bf = load_b_frag(Wo, DI, n0, ks, lane, 64);
    acc = wmma16(Af, Bf, acc);
  }
  int c = n0 + (lane & 15);
  int mb = (lane >> 4) << 3;
  int b = r0 / LSEQ;
  int l0 = r0 & (LSEQ - 1);
#pragma unroll
  for (int v = 0; v < 8; ++v) {
    int l = l0 + mb + v;
    size_t rix = ((size_t)(b * DM + c)) * LSEQ + l;
    float res = x0[rix] + x1[rix];
    fus[((size_t)(r0 + mb + v)) * DM + c] = (_Float16)(acc[v] + res);
  }
}

// =====================================================================
// K6: fc1 (N=256,K=64) + bias + exact gelu(erf). block=512, 16 rows/block
// =====================================================================
__global__ __launch_bounds__(512) void k_fc1(
    const _Float16* __restrict__ fus, const float* __restrict__ W,
    const float* __restrict__ bias, _Float16* __restrict__ h1 /*(B*L,256)*/) {
  __shared__ _Float16 asb[16][64];
  int r0 = blockIdx.x << 4;
  int tid = threadIdx.x;
  ((_Float16*)asb)[tid]       = fus[(size_t)r0 * DM + tid];
  ((_Float16*)asb)[tid + 512] = fus[(size_t)r0 * DM + tid + 512];
  __syncthreads();
  int w = tid >> 5, lane = tid & 31;
  int n0 = w << 4;
  v8f acc = {};
#pragma unroll
  for (int ks = 0; ks < 64; ks += 32) {
    v16h Af = load_a_frag(&asb[0][0], 64, ks, lane);
    v16h Bf = load_b_frag(W, 64, n0, ks, lane, 256);
    acc = wmma16(Af, Bf, acc);
  }
  int n = n0 + (lane & 15);
  int mb = (lane >> 4) << 3;
  float bv = bias[n];
#pragma unroll
  for (int v = 0; v < 8; ++v) {
    float x = acc[v] + bv;
    float ge = 0.5f * x * (1.f + erff(x * 0.70710678f));
    h1[((size_t)(r0 + mb + v)) * 256 + n] = (_Float16)ge;
  }
}

// =====================================================================
// K7: fc2 (N=768,K=256) + bias, fused final LayerNorm over 768 -> d_out
// block = 512 (16 waves x 3 col-tiles each), 16 rows/block
// =====================================================================
__global__ __launch_bounds__(512) void k_fc2_ln(
    const _Float16* __restrict__ h1, const float* __restrict__ W,
    const float* __restrict__ bias, const float* __restrict__ g,
    const float* __restrict__ be, float* __restrict__ out /*(B*L,768)*/) {
  __shared__ _Float16 asb[16][256];   // 8 KB
  __shared__ float    yrow[16][768];  // 48 KB (<= 320KB WGP LDS)
  int r0 = blockIdx.x << 4;
  int tid = threadIdx.x;
#pragma unroll
  for (int i = 0; i < 8; ++i)
    ((_Float16*)asb)[tid + i * 512] = h1[(size_t)r0 * 256 + tid + i * 512];
  __syncthreads();
  int w = tid >> 5, lane = tid & 31;
#pragma unroll
  for (int j = 0; j < 3; ++j) {
    int n0 = (w << 4) + j * 256;
    v8f acc = {};
#pragma unroll
    for (int ks = 0; ks < 256; ks += 32) {
      v16h Af = load_a_frag(&asb[0][0], 256, ks, lane);
      v16h Bf = load_b_frag(W, 256, n0, ks, lane, 768);
      acc = wmma16(Af, Bf, acc);
    }
    int n = n0 + (lane & 15);
    int mb = (lane >> 4) << 3;
    float bv = bias[n];
#pragma unroll
    for (int v = 0; v < 8; ++v) yrow[mb + v][n] = acc[v] + bv;
  }
  __syncthreads();
  int row = tid >> 5;
  float s = 0.f, sq = 0.f;
#pragma unroll
  for (int i = 0; i < 24; ++i) { float v = yrow[row][lane + i * 32]; s += v; sq += v * v; }
#pragma unroll
  for (int o = 16; o > 0; o >>= 1) { s += __shfl_xor(s, o, 32); sq += __shfl_xor(sq, o, 32); }
  float mean = s / 768.f;
  float rs = rsqrtf(sq / 768.f - mean * mean + 1e-5f);
  float* dst = out + (size_t)(r0 + row) * 768;
#pragma unroll
  for (int i = 0; i < 24; ++i) {
    int c = lane + i * 32;
    dst[c] = (yrow[row][c] - mean) * rs * g[c] + be[c];
  }
}

// =====================================================================
extern "C" void kernel_launch(void* const* d_in, const int* in_sizes, int n_in,
                              void* d_out, int out_size, void* d_ws, size_t ws_size,
                              hipStream_t stream) {
  (void)in_sizes; (void)n_in; (void)out_size; (void)ws_size;
  const float* x0   = (const float*)d_in[0];
  const float* x1   = (const float*)d_in[1];
  const float* ing  = (const float*)d_in[2];
  const float* inb  = (const float*)d_in[3];
  const float* W0   = (const float*)d_in[4];
  const float* W1   = (const float*)d_in[5];
  const float* xpw  = (const float*)d_in[6];
  const float* dtw  = (const float*)d_in[7];
  const float* dtb  = (const float*)d_in[8];
  const float* Alog = (const float*)d_in[9];
  const float* Dsv  = (const float*)d_in[10];
  const float* ong  = (const float*)d_in[11];
  const float* onb  = (const float*)d_in[12];
  const float* Wo   = (const float*)d_in[13];
  const float* fw1  = (const float*)d_in[14];
  const float* fb1  = (const float*)d_in[15];
  const float* fw2  = (const float*)d_in[16];
  const float* fb2  = (const float*)d_in[17];
  const float* lng  = (const float*)d_in[18];
  const float* lnb  = (const float*)d_in[19];
  float* out = (float*)d_out;

  // workspace carve-up (all sizes 256B aligned by construction)
  char* ws = (char*)d_ws;
  float* xbuf  = (float*)ws;                 ws += (size_t)NB * DI * LSEQ * 4;        // 8 MB
  float* zbuf  = (float*)ws;                 ws += (size_t)NB * LSEQ * DI * 4;        // 8 MB
  float* dtbuf = (float*)ws;                 ws += (size_t)NB * KD * LSEQ * DI * 4;   // 32 MB
  float* BCbuf = (float*)ws;                 ws += (size_t)NB * KD * LSEQ * 32 * 4;   // 8 MB
  float* ybuf  = (float*)ws;                 ws += (size_t)NB * KD * DI * LSEQ * 4;   // 32 MB
  _Float16* yz  = (_Float16*)ws;             ws += (size_t)NB * LSEQ * DI * 2;        // 4 MB
  _Float16* fus = (_Float16*)ws;             ws += (size_t)NB * LSEQ * DM * 2;        // 2 MB
  _Float16* h1  = (_Float16*)ws;             ws += (size_t)NB * LSEQ * 256 * 2;       // 8 MB

  const int rows = NB * LSEQ;  // 16384

  k_inproj <<<NB * (LSEQ / 16),      512, 0, stream>>>(x0, x1, ing, inb, W0, W1, xbuf, zbuf);
  k_xdbl   <<<NB * KD * (LSEQ / 16), 128, 0, stream>>>(xbuf, xpw, dtw, dtb, BCbuf, dtbuf);
  k_scan   <<<NB * KD * 4,           512, 0, stream>>>(xbuf, dtbuf, BCbuf, Alog, Dsv, ybuf);
  k_combine<<<NB * LSEQ,             128, 0, stream>>>(ybuf, zbuf, ong, onb, yz);
  k_outproj<<<rows / 16,             128, 0, stream>>>(yz, Wo, x0, x1, fus);
  k_fc1    <<<rows / 16,             512, 0, stream>>>(fus, fw1, fb1, h1);
  k_fc2_ln <<<rows / 16,             512, 0, stream>>>(h1, fw2, fb2, lng, lnb, out);
}
